// Dilated_attn_16131897164207
// MI455X (gfx1250) — matrix operand order repaired and checked by reading, not compiled, not run on hardware
//
#include <hip/hip_runtime.h>

#define DIM    512
#define HEADS  16
#define HD     32
#define NBATCH 1024
#define ATTN_SCALE 0.17677669529663687f   // 32^-0.5

typedef __attribute__((ext_vector_type(16))) _Float16 v16h;
typedef __attribute__((ext_vector_type(8)))  float    v8f;

// ---------------- LDS layout (units = ushort/f16) ----------------
#define XS_STRIDE 520                       // 512 + 8 pad
#define XS_OFF    0
#define XS_SIZE   (64 * XS_STRIDE)          // 33280
#define QS_STRIDE 40                        // 32 + 8 pad
#define QS_HEAD   (64 * QS_STRIDE)          // 2560
#define QS_OFF    (XS_OFF + XS_SIZE)
#define QS_SIZE   (HEADS * QS_HEAD)         // 40960
#define KS_OFF    (QS_OFF + QS_SIZE)
#define KS_SIZE   QS_SIZE
#define VT_STRIDE 72                        // 64 + 8 pad
#define VT_HEAD   (32 * VT_STRIDE)          // 2304
#define VT_OFF    (KS_OFF + KS_SIZE)
#define VT_SIZE   (HEADS * VT_HEAD)         // 36864
#define SMEM_USHORTS (VT_OFF + VT_SIZE)     // 152064
#define SMEM_BYTES   (SMEM_USHORTS * 2)     // 304128  (<= 320KB WGP LDS)
#define PS_OFF    XS_OFF                    // P scratch aliases X region (8 waves * 4096)

__device__ __forceinline__ unsigned short f2h(float f) {
  union { _Float16 h; unsigned short u; } c; c.h = (_Float16)f; return c.u;
}

// A fragment (16x32 f16): lane holds row (lane&15); K = kb+sel*8+0..7 (v0-3), kb+16+sel*8 (v4-7)
__device__ __forceinline__ v16h load_a_frag(const unsigned short* base, int stride,
                                            int row, int kb, int lane) {
  int sel = (lane >> 4) & 1;
  const unsigned short* p = base + row * stride + kb + sel * 8;
  union { v16h v; uint4 q[2]; } u;
  u.q[0] = *(const uint4*)p;
  u.q[1] = *(const uint4*)(p + 16);
  return u.v;
}
// B fragment (32x16 f16): lane holds column (lane&15); K = kb+sel*8+0..7 (v0-3), kb+16+sel*8 (v4-7)
__device__ __forceinline__ v16h load_b_frag(const unsigned short* base, int stride,
                                            int row, int kb, int lane) {
  int sel = (lane >> 4) & 1;
  const unsigned short* p = base + row * stride + kb + sel * 8;
  union { v16h v; uint4 q[2]; } u;
  u.q[0] = *(const uint4*)p;
  u.q[1] = *(const uint4*)(p + 16);
  return u.v;
}

__device__ __forceinline__ v8f wmma32(v16h a, v16h b, v8f c) {
  return __builtin_amdgcn_wmma_f32_16x16x32_f16(false, a, false, b, (short)0, c, false, false);
}

__global__ __launch_bounds__(256) void convert_weights(
    const float* __restrict__ qkv_w, const float* __restrict__ proj_w,
    unsigned short* __restrict__ qkvw_h, unsigned short* __restrict__ projw_h) {
  int i = blockIdx.x * 256 + threadIdx.x;
  if (i < 1536 * 512) qkvw_h[i] = f2h(qkv_w[i]);
  if (i < 512 * 512)  projw_h[i] = f2h(proj_w[i]);
}

__global__ __launch_bounds__(256) void fused_dilated_attn(
    const float* __restrict__ x, const float* __restrict__ qkv_b,
    const float* __restrict__ proj_b,
    const unsigned short* __restrict__ qkvw_h,
    const unsigned short* __restrict__ projw_h,
    float* __restrict__ out) {
  extern __shared__ unsigned short smem[];
  unsigned short* Xs = smem + XS_OFF;
  unsigned short* Qs = smem + QS_OFF;
  unsigned short* Ks = smem + KS_OFF;
  unsigned short* VT = smem + VT_OFF;

  const int tid  = threadIdx.x;
  const int lane = tid & 31;
  const int wave = tid >> 5;
  const int ln16 = lane & 15;
  const int hi16 = (lane >> 4) & 1;

  // window decode: batch = ((bb,wy,wx) window) * 4 + (dy,dx) dilation offset
  const int b   = blockIdx.x;
  const int dy  = (b >> 1) & 1;
  const int dx  = b & 1;
  const int win = b >> 2;
  const int wx  = win & 7;
  const int wy  = (win >> 3) & 7;
  const int bb  = win >> 6;

  // ---------- Phase A: gather 64 tokens x 512ch, f32 -> f16 into LDS ----------
  {
    const int m  = tid >> 2;
    const int c0 = (tid & 3) * 128;
    const int hh = wy * 16 + ((m >> 3) << 1) + dy;
    const int ww = wx * 16 + ((m & 7) << 1) + dx;
    const long gtok = ((long)bb * 128 + hh) * 128 + ww;
    const float4* src = (const float4*)(x + gtok * DIM + c0);
    unsigned int* dst = (unsigned int*)(Xs + m * XS_STRIDE + c0);
#pragma unroll 8
    for (int i = 0; i < 32; ++i) {
      float4 f = src[i];
      dst[i * 2]     = (unsigned)f2h(f.x) | ((unsigned)f2h(f.y) << 16);
      dst[i * 2 + 1] = (unsigned)f2h(f.z) | ((unsigned)f2h(f.w) << 16);
    }
  }
  __syncthreads();

  // ---------- Phase B: QKV GEMM 64x1536x512 + bias + RoPE epilogue ----------
  for (int t = wave; t < 4 * 96; t += 8) {
    const int mt = t / 96;
    const int nt = t % 96;
    const int j  = nt * 16 + ln16;            // output feature 0..1535
    const float bias = qkv_b[j];
    v8f acc = {bias, bias, bias, bias, bias, bias, bias, bias};
#pragma unroll 4
    for (int kb = 0; kb < DIM; kb += 32) {
      v16h a  = load_a_frag(Xs, XS_STRIDE, mt * 16 + ln16, kb, lane);
      v16h bw = load_b_frag(qkvw_h, DIM, j, kb, lane);
      acc = wmma32(a, bw, acc);
    }
    const int which = j >> 9;                 // 0=q 1=k 2=v
    const int rem   = j & 511;
    const int h     = rem >> 5;
    const int d     = rem & 31;
    if (which == 2) {
#pragma unroll
      for (int v = 0; v < 8; ++v) {
        int m = mt * 16 + v + (hi16 << 3);
        VT[h * VT_HEAD + d * VT_STRIDE + m] = f2h(acc[v]);   // store V transposed
      }
    } else {
      const int pidx    = (d & 15) >> 1;
      const int halfsel = d >> 4;             // 0: h-angle, 1: w-angle
      const float freq  = __expf(-1.1512925465f * (float)pidx); // 10000^(-p/8)
      unsigned short* dstb = (which == 0) ? (Qs + h * QS_HEAD) : (Ks + h * QS_HEAD);
#pragma unroll
      for (int v = 0; v < 8; ++v) {
        int m = mt * 16 + v + (hi16 << 3);
        float val = acc[v];
        float partner = __shfl_xor(val, 1, 32);               // RoPE pair across lanes
        float pos = halfsel ? (float)(m & 7) : (float)(m >> 3);
        float ang = pos * freq;
        float cs = __cosf(ang), sn = __sinf(ang);
        float r = (d & 1) ? (partner * sn + val * cs) : (val * cs - partner * sn);
        if (which == 0) r *= ATTN_SCALE;
        dstb[m * QS_STRIDE + d] = f2h(r);
      }
    }
  }
  __syncthreads();

  // ---------- Phase C: per-head attention (2 heads per wave) ----------
  for (int hh2 = 0; hh2 < 2; ++hh2) {
    const int h = wave * 2 + hh2;
    const unsigned short* Qh = Qs + h * QS_HEAD;
    const unsigned short* Kh = Ks + h * QS_HEAD;
    const unsigned short* Vh = VT + h * VT_HEAD;
    unsigned short* Ph = smem + PS_OFF + wave * 4096;   // 64x64 f16 P scratch

    v8f S[4][4];
#pragma unroll
    for (int mt = 0; mt < 4; ++mt) {
      v16h a = load_a_frag(Qh, QS_STRIDE, mt * 16 + ln16, 0, lane);
#pragma unroll
      for (int nt = 0; nt < 4; ++nt) {
        v16h bk = load_b_frag(Kh, QS_STRIDE, nt * 16 + ln16, 0, lane);
        v8f z = {0.f, 0.f, 0.f, 0.f, 0.f, 0.f, 0.f, 0.f};
        S[mt][nt] = wmma32(a, bk, z);
      }
    }
    // row softmax: row m lives in one 16-lane group at vgpr (m&7)
#pragma unroll
    for (int mt = 0; mt < 4; ++mt) {
#pragma unroll
      for (int v = 0; v < 8; ++v) {
        float mx = fmaxf(fmaxf(S[mt][0][v], S[mt][1][v]), fmaxf(S[mt][2][v], S[mt][3][v]));
#pragma unroll
        for (int off = 8; off >= 1; off >>= 1) mx = fmaxf(mx, __shfl_xor(mx, off, 32));
        float e0 = __expf(S[mt][0][v] - mx), e1 = __expf(S[mt][1][v] - mx);
        float e2 = __expf(S[mt][2][v] - mx), e3 = __expf(S[mt][3][v] - mx);
        float sum = e0 + e1 + e2 + e3;
#pragma unroll
        for (int off = 8; off >= 1; off >>= 1) sum += __shfl_xor(sum, off, 32);
        float rinv = 1.0f / sum;
        int row = mt * 16 + v + (hi16 << 3);
        unsigned short* pr = Ph + row * 64 + ln16;
        pr[0]  = f2h(e0 * rinv);
        pr[16] = f2h(e1 * rinv);
        pr[32] = f2h(e2 * rinv);
        pr[48] = f2h(e3 * rinv);
      }
    }
    asm volatile("s_wait_dscnt 0" ::: "memory");  // P stores visible before frag reloads
    // O = P @ V  (K=64 in two 32-steps), overwrite dead Q slot with O
    unsigned short* Oh = Qs + h * QS_HEAD;
#pragma unroll
    for (int mt = 0; mt < 4; ++mt) {
#pragma unroll
      for (int dt = 0; dt < 2; ++dt) {
        v8f acc = {0.f, 0.f, 0.f, 0.f, 0.f, 0.f, 0.f, 0.f};
#pragma unroll
        for (int kt = 0; kt < 2; ++kt) {
          v16h a  = load_a_frag(Ph, 64, mt * 16 + ln16, kt * 32, lane);
          v16h bv = load_b_frag(Vh, VT_STRIDE, dt * 16 + ln16, kt * 32, lane);
          acc = wmma32(a, bv, acc);
        }
#pragma unroll
        for (int v = 0; v < 8; ++v) {
          int m = mt * 16 + v + (hi16 << 3);
          Oh[m * QS_STRIDE + dt * 16 + ln16] = f2h(acc[v]);
        }
      }
    }
  }
  __syncthreads();

  // ---------- Phase D: proj GEMM 64x512x512 + bias, scatter to output ----------
  for (int t = wave; t < 4 * 32; t += 8) {
    const int mt = t >> 5;
    const int nt = t & 31;
    const int n  = nt * 16 + ln16;
    const float bias = proj_b[n];
    v8f acc = {bias, bias, bias, bias, bias, bias, bias, bias};
#pragma unroll 4
    for (int kb = 0; kb < DIM; kb += 32) {
      // A from head-blocked O in LDS: K chunk [kb,kb+32) stays inside head kb>>5
      const unsigned short* pa = Qs + (kb >> 5) * QS_HEAD + (mt * 16 + ln16) * QS_STRIDE + hi16 * 8;
      union { v16h v; uint4 q[2]; } ua;
      ua.q[0] = *(const uint4*)pa;
      ua.q[1] = *(const uint4*)(pa + 16);
      v16h bw = load_b_frag(projw_h, DIM, n, kb, lane);
      acc = wmma32(ua.v, bw, acc);
    }
#pragma unroll
    for (int v = 0; v < 8; ++v) {
      int m  = mt * 16 + v + (hi16 << 3);
      int hh = wy * 16 + ((m >> 3) << 1) + dy;
      int ww = wx * 16 + ((m & 7) << 1) + dx;
      long gtok = ((long)bb * 128 + hh) * 128 + ww;
      out[gtok * DIM + n] = acc[v];
    }
  }
}

extern "C" void kernel_launch(void* const* d_in, const int* in_sizes, int n_in,
                              void* d_out, int out_size, void* d_ws, size_t ws_size,
                              hipStream_t stream) {
  const float* x      = (const float*)d_in[0];
  const float* qkv_w  = (const float*)d_in[1];
  const float* qkv_b  = (const float*)d_in[2];
  const float* proj_w = (const float*)d_in[3];
  const float* proj_b = (const float*)d_in[4];
  unsigned short* qkvw_h  = (unsigned short*)d_ws;        // 1536*512 f16
  unsigned short* projw_h = qkvw_h + 1536 * 512;          // 512*512 f16
  float* out = (float*)d_out;

  hipFuncSetAttribute(reinterpret_cast<const void*>(fused_dilated_attn),
                      hipFuncAttributeMaxDynamicSharedMemorySize, SMEM_BYTES);

  convert_weights<<<(1536 * 512 + 255) / 256, 256, 0, stream>>>(qkv_w, proj_w, qkvw_h, projw_h);
  fused_dilated_attn<<<NBATCH, 256, SMEM_BYTES, stream>>>(x, qkv_b, proj_b, qkvw_h, projw_h, out);
}
